// LSTM_38044820308379
// MI455X (gfx1250) — matrix operand order repaired and checked by reading, not compiled, not run on hardware
//
#include <hip/hip_runtime.h>
#include <stdint.h>

// ---------------------------------------------------------------------------
// LSTM scan, MI455X (gfx1250, wave32, WMMA).
//   T=512, B=64, D=1024, L=1024.  gates = [x_t, h] @ Wcat  (M=64,K=2048,N=4096)
//   * Wt f16, N-major, gate-interleaved (f,i,o,g adjacent columns) -> 16MB,
//     L2-resident across all 512 sequential steps.
//   * A operand (z = [x_t | h], f16) staged through LDS in double-buffered
//     64x128 K-chunks via GLOBAL_LOAD_ASYNC_TO_LDS_B128 (ASYNCcnt-tracked)
//     to kill the 8x per-wave load redundancy; padded LDS rows avoid bank
//     conflicts on the row-strided fragment reads.
//   * Per K-step: batch-load 4 A fragments (one DS clause) then issue the 4
//     WMMAs back-to-back; B fragments preloaded per chunk.
// ---------------------------------------------------------------------------

typedef __attribute__((ext_vector_type(16))) _Float16 v16h;
typedef __attribute__((ext_vector_type(8)))  _Float16 v8h;
typedef __attribute__((ext_vector_type(8)))  float    v8f;

#define T_STEPS 512
#define B_SZ    64
#define D_SZ    1024
#define L_SZ    1024
#define K_SZ    2048    // D + L
#define N_SZ    4096    // 4*L
#define NT      128     // N columns per workgroup (32 latents * 4 gates)
#define KC      128     // K per staged LDS chunk
#define NCHUNK  (K_SZ / KC)   // 16
#define APAD    136     // padded halves per LDS row (272B -> bank-conflict free)
#define ATILE   (B_SZ * APAD) // halves per A buffer

// ---- CDNA5 async global->LDS path -----------------------------------------
#if __has_builtin(__builtin_amdgcn_global_load_async_to_lds_b128)
#define HAVE_ASYNC_COPY 1
typedef int v4i_t __attribute__((vector_size(16)));
typedef __attribute__((address_space(1))) v4i_t* gv4i_p;
typedef __attribute__((address_space(3))) v4i_t* lv4i_p;
#else
#define HAVE_ASYNC_COPY 0
#endif

__device__ __forceinline__ void async_copy_wait0() {
#if HAVE_ASYNC_COPY
#if __has_builtin(__builtin_amdgcn_s_wait_asynccnt)
    __builtin_amdgcn_s_wait_asynccnt(0);
#else
    asm volatile("s_wait_asynccnt 0" ::: "memory");
#endif
#endif
}

// ---- fragment loader: 16-bit A/B 16x32 layout (ISA 7.12.2) -----------------
// lane l: row l&15, halves {koff..koff+7, koff+16..koff+23}, koff=(l>>4)*8
__device__ __forceinline__ v16h load_frag16(const _Float16* p) {
    v8h lo = *reinterpret_cast<const v8h*>(p);
    v8h hi = *reinterpret_cast<const v8h*>(p + 16);
    v16h r;
#pragma unroll
    for (int i = 0; i < 8; ++i) { r[i] = lo[i]; r[i + 8] = hi[i]; }
    return r;
}

__device__ __forceinline__ float fast_sigmoid(float x) {
    return 1.0f / (1.0f + __expf(-x));
}
__device__ __forceinline__ float fast_tanh(float x) {
    return 2.0f / (1.0f + __expf(-2.0f * x)) - 1.0f;
}

// ---- weight prep: Wt[4*j+g][k] = W_g[k][j] as f16 --------------------------
__global__ __launch_bounds__(256) void lstm_prep_w(
    const float* __restrict__ Wf, const float* __restrict__ Wi,
    const float* __restrict__ Wo, const float* __restrict__ Wg,
    _Float16* __restrict__ Wt)
{
    int id = blockIdx.x * 256 + threadIdx.x;     // N_SZ*K_SZ = 8388608 total
    int r  = id >> 11;                            // N row (0..4095)
    int k  = id & (K_SZ - 1);                     // K (0..2047)
    int j  = r >> 2;
    int g  = r & 3;
    const float* W = (g == 0) ? Wf : (g == 1) ? Wi : (g == 2) ? Wo : Wg;
    Wt[id] = (_Float16)W[k * L_SZ + j];
}

// ---- init: c = 0, h-half of z0 = 0 -----------------------------------------
__global__ __launch_bounds__(256) void lstm_init(
    float* __restrict__ c_ws, _Float16* __restrict__ z0)
{
    int id = blockIdx.x * 256 + threadIdx.x;     // 65536 = B*L
    c_ws[id] = 0.0f;
    int b = id >> 10, j = id & (L_SZ - 1);
    z0[b * K_SZ + D_SZ + j] = (_Float16)0.0f;
}

// ---- per-step: convert x_t (f32) into x-half of the active z buffer --------
__global__ __launch_bounds__(256) void lstm_cvt_x(
    const float* __restrict__ xt, _Float16* __restrict__ zin)
{
    int id = blockIdx.x * 256 + threadIdx.x;     // 65536 = B*D
    int b = id >> 10, k = id & (D_SZ - 1);
    zin[b * K_SZ + k] = (_Float16)xt[id];
}

// ---- stage one 64xKC z-chunk into LDS (async global->LDS) ------------------
__device__ __forceinline__ void stage_chunk(
    const _Float16* __restrict__ zin, _Float16* lbuf, int c, int tid)
{
#pragma unroll
    for (int i = 0; i < 4; ++i) {
        int u   = tid + i * 256;     // 0..1023 16-byte units
        int row = u >> 4;            // 0..63
        int q   = u & 15;            // 16B unit within row
        const _Float16* g = zin + (size_t)row * K_SZ + c * KC + q * 8;
        _Float16*       l = lbuf + row * APAD + q * 8;
#if HAVE_ASYNC_COPY
        __builtin_amdgcn_global_load_async_to_lds_b128(
            (gv4i_p)(uintptr_t)g, (lv4i_p)(uintptr_t)l, 0, 0);
#else
        *reinterpret_cast<v8h*>(l) = *reinterpret_cast<const v8h*>(g);
#endif
    }
}

// ---- fused step: gates = z @ Wt^T (WMMA), then cell update -----------------
// grid = 32 blocks (N tiles of 128), block = 256 threads = 8 waves.
// wave w owns rows 0..63 x cols [w*16, w*16+16): 4 accumulator tiles.
__global__ __launch_bounds__(256) void lstm_step(
    const _Float16* __restrict__ Wt,   // [N_SZ][K_SZ] gate-interleaved
    const _Float16* __restrict__ zin,  // [B][K_SZ]  = [x_t | h]
    _Float16*       __restrict__ zout, // next step's z buffer (h-half)
    float*          __restrict__ c_ws, // [B][L]
    const float* __restrict__ bf, const float* __restrict__ bi,
    const float* __restrict__ bo, const float* __restrict__ bg,
    float*          __restrict__ out_t) // [B][L] -> hs[t]
{
    // A double-buffer (2 x 17408B); gate buffer (32KB) aliases it afterwards.
    __shared__ __align__(16) char smem[2 * ATILE * sizeof(_Float16)];
    _Float16* atile = reinterpret_cast<_Float16*>(smem);
    float*    gate  = reinterpret_cast<float*>(smem);

    const int tid  = threadIdx.x;
    const int nt   = blockIdx.x;                 // 0..31
    const int lane = tid & 31;
    const int wave = tid >> 5;                   // 0..7

    const int hrow = lane & 15;
    const int koff = (lane >> 4) * 8;

    const int ncol = nt * NT + wave * 16 + hrow; // B column owned by this lane
    const _Float16* bbase = Wt + (size_t)ncol * K_SZ + koff;

    v8f acc[4];
#pragma unroll
    for (int m = 0; m < 4; ++m) acc[m] = (v8f){};

    stage_chunk(zin, atile, 0, tid);
    async_copy_wait0();
    __syncthreads();

    for (int c = 0; c < NCHUNK; ++c) {
        const _Float16* abuf = atile + (c & 1) * ATILE;

        // preload the whole chunk's B fragments (global, L2-resident)
        v16h bfr[4];
#pragma unroll
        for (int s = 0; s < 4; ++s)
            bfr[s] = load_frag16(bbase + c * KC + s * 32);

        // kick off next chunk's A staging while we compute
        if (c + 1 < NCHUNK)
            stage_chunk(zin, atile + ((c + 1) & 1) * ATILE, c + 1, tid);

#pragma unroll
        for (int s = 0; s < 4; ++s) {
            // batch all 4 A fragments (8 ds_load_b128, one clause, one wait)
            v16h af[4];
#pragma unroll
            for (int m = 0; m < 4; ++m)
                af[m] = load_frag16(abuf + (m * 16 + hrow) * APAD + s * 32 + koff);
            // then 4 back-to-back WMMAs
#pragma unroll
            for (int m = 0; m < 4; ++m)
                acc[m] = __builtin_amdgcn_wmma_f32_16x16x32_f16(
                    false, af[m], false, bfr[s], (short)0, acc[m], false, false);
        }

        async_copy_wait0();   // next chunk staged in LDS
        __syncthreads();      // everyone done reading abuf before it's reused
    }

    // spill D tiles to (aliased) gate buffer
#pragma unroll
    for (int m = 0; m < 4; ++m) {
        const int col = wave * 16 + (lane & 15);
        const int rb  = m * 16 + (lane >> 4) * 8;
#pragma unroll
        for (int v = 0; v < 8; ++v)
            gate[(rb + v) * NT + col] = acc[m][v];
    }
    __syncthreads();

    // elementwise cell update: 2048 (b, j) cells per block, 8 per thread
#pragma unroll
    for (int i = 0; i < 8; ++i) {
        const int idx = tid + i * 256;           // 0..2047
        const int b   = idx >> 5;                // 0..63
        const int jl  = idx & 31;                // local latent
        const int jg  = nt * 32 + jl;            // global latent

        const float* gp = &gate[b * NT + 4 * jl];
        float fg = fast_sigmoid(gp[0] + bf[jg]);
        float ig = fast_sigmoid(gp[1] + bi[jg]);
        float og = fast_sigmoid(gp[2] + bo[jg]);
        float gg = fast_tanh   (gp[3] + bg[jg]);

        const int ci = b * L_SZ + jg;
        float cn = fg * c_ws[ci] + ig * gg;
        c_ws[ci] = cn;
        float hn = og * fast_tanh(cn);

        out_t[ci] = hn;                               // hs[t] (f32)
        zout[b * K_SZ + D_SZ + jg] = (_Float16)hn;    // h for step t+1
    }
}

// ---------------------------------------------------------------------------
extern "C" void kernel_launch(void* const* d_in, const int* in_sizes, int n_in,
                              void* d_out, int out_size, void* d_ws, size_t ws_size,
                              hipStream_t stream) {
    (void)in_sizes; (void)n_in; (void)out_size; (void)ws_size;

    const float* x  = (const float*)d_in[0];
    const float* Wf = (const float*)d_in[1];
    const float* bf = (const float*)d_in[2];
    const float* Wi = (const float*)d_in[3];
    const float* bi = (const float*)d_in[4];
    const float* Wo = (const float*)d_in[5];
    const float* bo = (const float*)d_in[6];
    const float* Wg = (const float*)d_in[7];
    const float* bg = (const float*)d_in[8];
    float* out = (float*)d_out;

    // workspace layout (<= 17 MB)
    char* ws = (char*)d_ws;
    _Float16* Wt  = (_Float16*)(ws);                                 // 16 MB
    _Float16* z0  = (_Float16*)(ws + (16u << 20));                   // 256 KB
    _Float16* z1  = (_Float16*)(ws + (16u << 20) + (256u << 10));    // 256 KB
    float*    cws = (float*)   (ws + (16u << 20) + (512u << 10));    // 256 KB

    lstm_prep_w<<<(N_SZ * K_SZ) / 256, 256, 0, stream>>>(Wf, Wi, Wo, Wg, Wt);
    lstm_init<<<(B_SZ * L_SZ) / 256, 256, 0, stream>>>(cws, z0);

    for (int t = 0; t < T_STEPS; ++t) {
        _Float16* zin  = (t & 1) ? z1 : z0;
        _Float16* zout = (t & 1) ? z0 : z1;
        lstm_cvt_x<<<(B_SZ * D_SZ) / 256, 256, 0, stream>>>(
            x + (size_t)t * B_SZ * D_SZ, zin);
        lstm_step<<<N_SZ / NT, 256, 0, stream>>>(
            Wt, zin, zout, cws, bf, bi, bo, bg,
            out + (size_t)t * B_SZ * L_SZ);
    }
}
